// S3RNN_29755533427636
// MI455X (gfx1250) — compile-verified, matrix operand order checked by reading, not run on hardware
//
#include <hip/hip_runtime.h>
#include <hip/hip_bf16.h>

typedef __attribute__((ext_vector_type(16))) _Float16 v16h;
typedef __attribute__((ext_vector_type(8)))  _Float16 v8h;
typedef __attribute__((ext_vector_type(8)))  float    v8f;

#define S3_B     256
#define S3_T     2048
#define S3_V     64
#define S3_H     64
#define S3_EOS   2
#define SEG_LEN  64
#define NSEG     32              // T / SEG_LEN
#define NROWS    (S3_B * S3_T)   // 524288
#define LOGITS_N (NROWS * S3_V)  // 33554432

__device__ __forceinline__ float s3_gelu(float x) {
    return 0.5f * x * (1.0f + erff(x * 0.70710678118654752440f));
}

__device__ __forceinline__ void s3_qmul(float w1, float x1, float y1, float z1,
                                        float w2, float x2, float y2, float z2,
                                        float& w, float& x, float& y, float& z) {
    w = w1 * w2 - x1 * x2 - y1 * y2 - z1 * z2;
    x = w1 * x2 + x1 * w2 + y1 * z2 - z1 * y2;
    y = w1 * y2 - x1 * z2 + y1 * w2 + z1 * x2;
    z = w1 * z2 + x1 * y2 - y1 * x2 + z1 * w2;
}

__device__ __forceinline__ void s3_norm(float& w, float& x, float& y, float& z) {
    float n = sqrtf(w * w + x * x + y * y + z * z);
    float inv = 1.0f / fmaxf(n, 1e-12f);
    w *= inv; x *= inv; y *= inv; z *= inv;
}

// ---------------------------------------------------------------------------
// Phase 0: g depends only on the token -> 64-entry table.
// gtab[v] = normalize(gelu(eW1[v]+eb1) @ eW2 + eb2); EOS -> identity.
// ---------------------------------------------------------------------------
__global__ void s3_gtable(const float* __restrict__ eW1, const float* __restrict__ eb1,
                          const float* __restrict__ eW2, const float* __restrict__ eb2,
                          float* __restrict__ gtab) {
    int v = threadIdx.x;  // 0..63
    float a0 = eb2[0], a1 = eb2[1], a2 = eb2[2], a3 = eb2[3];
    for (int k = 0; k < S3_H; ++k) {
        float h = s3_gelu(eW1[v * S3_H + k] + eb1[k]);
        a0 += h * eW2[k * 4 + 0];
        a1 += h * eW2[k * 4 + 1];
        a2 += h * eW2[k * 4 + 2];
        a3 += h * eW2[k * 4 + 3];
    }
    s3_norm(a0, a1, a2, a3);
    if (v == S3_EOS) { a0 = 1.0f; a1 = 0.0f; a2 = 0.0f; a3 = 0.0f; }
    gtab[v * 4 + 0] = a0; gtab[v * 4 + 1] = a1;
    gtab[v * 4 + 2] = a2; gtab[v * 4 + 3] = a3;
}

// ---------------------------------------------------------------------------
// Phase A: segment-local normalized quaternion prefix products.
// One thread per (row, segment): 8192 chains of 64 steps.
// ---------------------------------------------------------------------------
__global__ __launch_bounds__(256) void s3_seg(const int* __restrict__ tokens,
                                              const float* __restrict__ gtab,
                                              float* __restrict__ localC) {
    __shared__ float gt[S3_V * 4];
    gt[threadIdx.x] = gtab[threadIdx.x];
    __syncthreads();
    int id = blockIdx.x * 256 + threadIdx.x;   // 0..8191
    int b  = id >> 5;                           // wave == one batch row
    int s  = id & 31;
    int t0 = s * SEG_LEN;
    const int* tp = tokens + b * S3_T + t0;
    float*     cp = localC + (size_t)(b * S3_T + t0) * 4;
    float cw = 1.0f, cx = 0.0f, cy = 0.0f, cz = 0.0f;
    for (int i = 0; i < SEG_LEN; ++i) {
        int tok = tp[i];
        float gw = gt[tok * 4 + 0], gx = gt[tok * 4 + 1];
        float gy = gt[tok * 4 + 2], gz = gt[tok * 4 + 3];
        float w, x, y, z;
        s3_qmul(cw, cx, cy, cz, gw, gx, gy, gz, w, x, y, z);
        s3_norm(w, x, y, z);
        cw = w; cx = x; cy = y; cz = z;
        float4 o; o.x = cw; o.y = cx; o.z = cy; o.w = cz;
        *(float4*)(cp + i * 4) = o;
    }
}

// ---------------------------------------------------------------------------
// Phase B: exclusive scan of segment totals per row (tiny: 256 threads x 32).
// ---------------------------------------------------------------------------
__global__ void s3_scan(const float* __restrict__ localC, float* __restrict__ segOff) {
    int b = threadIdx.x;  // 0..255
    float ow = 1.0f, ox = 0.0f, oy = 0.0f, oz = 0.0f;
    for (int s = 0; s < NSEG; ++s) {
        float4 o; o.x = ow; o.y = ox; o.z = oy; o.w = oz;
        *(float4*)(segOff + (size_t)(b * NSEG + s) * 4) = o;
        const float* tp = localC + (size_t)(b * S3_T + s * SEG_LEN + SEG_LEN - 1) * 4;
        float w, x, y, z;
        s3_qmul(ow, ox, oy, oz, tp[0], tp[1], tp[2], tp[3], w, x, y, z);
        s3_norm(w, x, y, z);
        ow = w; ox = x; oy = y; oz = z;
    }
}

// ---------------------------------------------------------------------------
// Phase C: fused fixup + sigma + head MLP.  256 thr (8 waves) per block,
// 16 rows per wave, 128 rows per block, 4096 blocks.
// GEMM1 (K=4) on VALU; GEMM2 (64x64, K=64) via v_wmma_f32_16x16x32_f16.
// LDS rows padded to 72 halves (36 dwords) -> conflict-free ds_load_b128.
// ---------------------------------------------------------------------------
__global__ __launch_bounds__(256) void s3_head(const float* __restrict__ localC,
                                               const float* __restrict__ segOff,
                                               const float* __restrict__ hW1,
                                               const float* __restrict__ hb1,
                                               const float* __restrict__ hW2,
                                               const float* __restrict__ hb2,
                                               float* __restrict__ out) {
    __shared__ float sW1[4 * 64];
    __shared__ float sb1[64];
    __shared__ float sb2[64];
    __shared__ __align__(16) _Float16 sW2T[64 * 72];      // hW2^T, padded pitch
    __shared__ __align__(16) _Float16 sA[8 * 16 * 72];    // per-wave X tile (f16)
    __shared__ float sC[8 * 16 * 4];                      // per-wave quaternions

    int tid = threadIdx.x, wave = tid >> 5, lane = tid & 31;

    sW1[tid] = hW1[tid];                                  // 256 floats
    if (tid < 64) { sb1[tid] = hb1[tid]; sb2[tid] = hb2[tid]; }
    for (int i = tid; i < 64 * 64; i += 256) {            // transpose hW2 -> f16
        int k = i >> 6, n = i & 63;
        sW2T[n * 72 + k] = (_Float16)hW2[i];
    }
    __syncthreads();

    long r0 = (long)blockIdx.x * 128 + wave * 16;         // first global row of tile
    int  b  = (int)(r0 >> 11);
    int  t0 = (int)(r0 & 2047);

    if (lane < 16) {                                      // full C + sigma per row
        int t = t0 + lane;
        const float* lc = localC + (r0 + lane) * 4;
        const float* of = segOff + (size_t)(b * NSEG + (t >> 6)) * 4;
        float w, x, y, z;
        s3_qmul(of[0], of[1], of[2], of[3], lc[0], lc[1], lc[2], lc[3], w, x, y, z);
        s3_norm(w, x, y, z);
        sC[wave * 64 + lane * 4 + 0] = w;
        sC[wave * 64 + lane * 4 + 1] = x;
        sC[wave * 64 + lane * 4 + 2] = y;
        sC[wave * 64 + lane * 4 + 3] = z;
        out[(long)LOGITS_N + r0 + lane] = acosf(fminf(fabsf(w), 1.0f - 1e-07f));
    }
    __syncthreads();

    // GEMM1 + gelu -> f16 A tile in LDS (16 rows x 64 cols per wave)
    for (int i = lane; i < 16 * 64; i += 32) {
        int r = i >> 6, c = i & 63;
        float Cw = sC[wave * 64 + r * 4 + 0];
        float Cx = sC[wave * 64 + r * 4 + 1];
        float Cy = sC[wave * 64 + r * 4 + 2];
        float Cz = sC[wave * 64 + r * 4 + 3];
        float x = Cw * sW1[c] + Cx * sW1[64 + c] + Cy * sW1[128 + c] + Cz * sW1[192 + c] + sb1[c];
        sA[wave * (16 * 72) + r * 72 + c] = (_Float16)s3_gelu(x);
    }
    __syncthreads();

    // A operands per ISA 16-bit 16x32 layout: lane holds row (lane&15),
    // K-octet (lane>>4)*8 in V0-3 and +16 in V4-7.
    int row = lane & 15;
    int kq  = (lane >> 4) * 8;
    const _Float16* ab = &sA[wave * (16 * 72) + row * 72 + kq];
    v8h a0l = *(const v8h*)(ab);        v8h a0h = *(const v8h*)(ab + 16);
    v8h a1l = *(const v8h*)(ab + 32);   v8h a1h = *(const v8h*)(ab + 48);
    v16h A0 = __builtin_shufflevector(a0l, a0h, 0,1,2,3,4,5,6,7,8,9,10,11,12,13,14,15);
    v16h A1 = __builtin_shufflevector(a1l, a1h, 0,1,2,3,4,5,6,7,8,9,10,11,12,13,14,15);

    long rbase = r0 + ((lane >> 4) * 8);
    for (int j = 0; j < 4; ++j) {
        int n0 = j * 16;
        const _Float16* bb = &sW2T[(n0 + row) * 72 + kq];
        v8h b0l = *(const v8h*)(bb);       v8h b0h = *(const v8h*)(bb + 16);
        v8h b1l = *(const v8h*)(bb + 32);  v8h b1h = *(const v8h*)(bb + 48);
        v16h B0 = __builtin_shufflevector(b0l, b0h, 0,1,2,3,4,5,6,7,8,9,10,11,12,13,14,15);
        v16h B1 = __builtin_shufflevector(b1l, b1h, 0,1,2,3,4,5,6,7,8,9,10,11,12,13,14,15);
        float bz = sb2[n0 + row];
        v8f acc = {bz, bz, bz, bz, bz, bz, bz, bz};
        acc = __builtin_amdgcn_wmma_f32_16x16x32_f16(false, A0, false, B0, (short)0, acc, false, false);
        acc = __builtin_amdgcn_wmma_f32_16x16x32_f16(false, A1, false, B1, (short)0, acc, false, false);
        int col = n0 + row;
        #pragma unroll
        for (int vv = 0; vv < 8; ++vv)
            out[(rbase + vv) * 64 + col] = acc[vv];
    }
}

extern "C" void kernel_launch(void* const* d_in, const int* in_sizes, int n_in,
                              void* d_out, int out_size, void* d_ws, size_t ws_size,
                              hipStream_t stream) {
    const int*   tokens = (const int*)  d_in[0];
    const float* eW1    = (const float*)d_in[1];
    const float* eb1    = (const float*)d_in[2];
    const float* eW2    = (const float*)d_in[3];
    const float* eb2    = (const float*)d_in[4];
    const float* hW1    = (const float*)d_in[5];
    const float* hb1    = (const float*)d_in[6];
    const float* hW2    = (const float*)d_in[7];
    const float* hb2    = (const float*)d_in[8];
    float* out = (float*)d_out;

    float* ws     = (float*)d_ws;
    float* localC = ws;                          // B*T*4   = 2,097,152 f
    float* segOff = ws + (size_t)NROWS * 4;      // B*32*4  =    32,768 f
    float* gtab   = segOff + S3_B * NSEG * 4;    // 64*4    =       256 f

    s3_gtable<<<1, 64, 0, stream>>>(eW1, eb1, eW2, eb2, gtab);
    s3_seg<<<(S3_B * NSEG) / 256, 256, 0, stream>>>(tokens, gtab, localC);
    s3_scan<<<1, 256, 0, stream>>>(localC, segOff);
    s3_head<<<NROWS / 128, 256, 0, stream>>>(localC, segOff, hW1, hb1, hW2, hb2, out);
}
